// MultiHeadAttention_5549097747188
// MI455X (gfx1250) — compile-verified, matrix operand order checked by reading, not run on hardware
//
#include <hip/hip_runtime.h>
#include <math.h>

typedef __bf16 bf16;
typedef __attribute__((ext_vector_type(16))) __bf16 v16bf;
typedef __attribute__((ext_vector_type(8)))  __bf16 v8bf;
typedef __attribute__((ext_vector_type(8)))  float  v8f;

union Frag {
  v16bf v;
  v8bf  h[2];
};

__device__ __forceinline__ unsigned short bf16_bits(float f) {
  unsigned u = __float_as_uint(f);
  u += 0x7FFFu + ((u >> 16) & 1u);   // round-to-nearest-even
  return (unsigned short)(u >> 16);
}
__device__ __forceinline__ bf16 to_bf16(float f) {
  return __builtin_bit_cast(bf16, bf16_bits(f));
}

__device__ __forceinline__ v8f wmma_bf16(const Frag& a, const Frag& b, v8f c) {
  return __builtin_amdgcn_wmma_f32_16x16x32_bf16(
      /*neg_a=*/false, a.v, /*neg_b=*/false, b.v,
      /*c_mod=*/(short)0, c, /*reuse_a=*/false, /*reuse_b=*/false);
}

__device__ __forceinline__ void load_frag(Frag& f, const bf16* p) {
  f.h[0] = *(const v8bf*)(p);
  f.h[1] = *(const v8bf*)(p + 16);
}

// ---------------------------------------------------------------- convert (4 f32 -> 4 bf16 per thread)
struct us4 { unsigned short x, y, z, w; };
__global__ void mha_cvt_bf16(const float* __restrict__ x, bf16* __restrict__ y, int n4) {
  int i = blockIdx.x * blockDim.x + threadIdx.x;
  if (i < n4) {
    float4 f = ((const float4*)x)[i];
    us4 o;
    o.x = bf16_bits(f.x); o.y = bf16_bits(f.y);
    o.z = bf16_bits(f.z); o.w = bf16_bits(f.w);
    ((us4*)y)[i] = o;
  }
}

// ---------------------------------------------------------------- GEMM  Y = X @ W^T + bias
// Each wave computes a 16x64 tile; software-pipelined (ping-pong) K loop.
// X:[M][512] bf16, W:[512][512] bf16 (row n holds k), bias:[512] f32
// mode 0: Yb row-major bf16 [M][512]
// mode 1: Yb "V-transposed" bf16: row (m/4096)*512 + n, col m%4096, ld=4096
// mode 2: Yf row-major f32 [M][512]
__global__ void mha_gemm_xwt(const bf16* __restrict__ X, const bf16* __restrict__ W,
                             const float* __restrict__ bias,
                             bf16* __restrict__ Yb, float* __restrict__ Yf,
                             int M, int mode) {
  const int wave = threadIdx.x >> 5;
  const int lane = threadIdx.x & 31;
  const int half = (lane >> 4) & 1;
  const int ln   = lane & 15;
  const int klo  = half ? 8 : 0;
  const int tile = blockIdx.x * 4 + wave;      // blockDim = 128 (4 waves)
  const int m0 = (tile >> 3) * 16;             // 8 n-tiles of width 64
  const int n0 = (tile & 7) * 64;
  if (m0 >= M) return;

  const bf16* Xr = X + (size_t)(m0 + ln) * 512 + klo;
  const bf16* Wr = W + (size_t)(n0 + ln) * 512 + klo;   // +t*16*512 per sub-tile
  v8f c0 = {}, c1 = {}, c2 = {}, c3 = {};

  Frag aA, b0A, b1A, b2A, b3A;                 // stage A
  Frag aB, b0B, b1B, b2B, b3B;                 // stage B
  load_frag(aA,  Xr);
  load_frag(b0A, Wr);
  load_frag(b1A, Wr + 16 * 512);
  load_frag(b2A, Wr + 32 * 512);
  load_frag(b3A, Wr + 48 * 512);
  #pragma unroll 1
  for (int kk = 0; kk < 512; kk += 64) {
    const int kb = kk + 32;                    // stage-B loads in flight during A WMMAs
    load_frag(aB,  Xr + kb);
    load_frag(b0B, Wr + kb);
    load_frag(b1B, Wr + 16 * 512 + kb);
    load_frag(b2B, Wr + 32 * 512 + kb);
    load_frag(b3B, Wr + 48 * 512 + kb);
    c0 = wmma_bf16(aA, b0A, c0);
    c1 = wmma_bf16(aA, b1A, c1);
    c2 = wmma_bf16(aA, b2A, c2);
    c3 = wmma_bf16(aA, b3A, c3);
    const int ka = (kk + 64 < 512) ? kk + 64 : 0;  // clamped (redundant last load, in-bounds)
    load_frag(aA,  Xr + ka);
    load_frag(b0A, Wr + ka);
    load_frag(b1A, Wr + 16 * 512 + ka);
    load_frag(b2A, Wr + 32 * 512 + ka);
    load_frag(b3A, Wr + 48 * 512 + ka);
    c0 = wmma_bf16(aB, b0B, c0);
    c1 = wmma_bf16(aB, b1B, c1);
    c2 = wmma_bf16(aB, b2B, c2);
    c3 = wmma_bf16(aB, b3B, c3);
  }
  #pragma unroll
  for (int t = 0; t < 4; ++t) {
    const v8f& c = (t == 0) ? c0 : (t == 1) ? c1 : (t == 2) ? c2 : c3;
    const int n = n0 + t * 16 + ln;
    const float bv = bias[n];
    #pragma unroll
    for (int j = 0; j < 8; ++j) {
      const int m = m0 + j + half * 8;
      const float val = c[j] + bv;
      if (mode == 0) {
        Yb[(size_t)m * 512 + n] = to_bf16(val);
      } else if (mode == 1) {
        const int batch = m >> 12, s = m & 4095;       // S = 4096
        Yb[((size_t)(batch * 512 + n) << 12) + s] = to_bf16(val);
      } else {
        Yf[(size_t)m * 512 + n] = val;
      }
    }
  }
}

// ---------------------------------------------------------------- flash attention (64-key tiles)
// Qh,Kh: [B*S][512] bf16 (head h = cols h*64..h*64+63)
// Vt:    [B*512][4096] bf16 (row b*512 + h*64 + d, col s)
// ctx:   [B*S][512] bf16
__global__ void mha_flash_attn(const bf16* __restrict__ Qh, const bf16* __restrict__ Kh,
                               const bf16* __restrict__ Vt, bf16* __restrict__ ctx) {
  __shared__ bf16 Plds[8][16 * 64];                  // 2 KB per wave
  const int wave = threadIdx.x >> 5;
  const int lane = threadIdx.x & 31;
  const int half = (lane >> 4) & 1;
  const int ln   = lane & 15;
  const int klo  = half ? 8 : 0;

  const int wtile = blockIdx.x * 8 + wave;           // 4096 q-tiles total
  const int b  = wtile >> 11;                        // / (8 heads * 256 tiles)
  const int h  = (wtile >> 8) & 7;
  const int q0 = (wtile & 255) * 16;

  // Q fragments (d-steps 0 and 32), reused for all key tiles
  const bf16* Qr = Qh + (size_t)(b * 4096 + q0 + ln) * 512 + h * 64 + klo;
  Frag qa0, qa1;
  load_frag(qa0, Qr);
  load_frag(qa1, Qr + 32);

  float Mx[8], l[8];
  #pragma unroll
  for (int j = 0; j < 8; ++j) { Mx[j] = -1e30f; l[j] = 0.f; }
  v8f O0 = {}, O1 = {}, O2 = {}, O3 = {};

  const bf16* Kbase = Kh + (size_t)(b * 4096) * 512 + h * 64 + klo;
  const bf16* Vbase = Vt + ((size_t)(b * 512 + h * 64) << 12) + klo;
  const float scale = 0.125f;                        // 1/sqrt(64)

  for (int k0 = 0; k0 < 4096; k0 += 64) {
    if (k0 + 64 < 4096) {                            // -> global_prefetch_b8
      __builtin_prefetch(Kbase + (size_t)(k0 + 64 + ln) * 512, 0, 1);
      __builtin_prefetch(Vbase + ((size_t)ln << 12) + k0 + 64, 0, 1);
    }
    // ---- scores S = Q K^T  (16 q x 64 k), 8 WMMAs
    v8f c0 = {}, c1 = {}, c2 = {}, c3 = {};
    {
      const bf16* Kr0 = Kbase + (size_t)(k0 + ln) * 512;
      const bf16* Kr1 = Kr0 + (size_t)16 * 512;
      const bf16* Kr2 = Kr0 + (size_t)32 * 512;
      const bf16* Kr3 = Kr0 + (size_t)48 * 512;
      Frag k0a, k0b, k1a, k1b, k2a, k2b, k3a, k3b;
      load_frag(k0a, Kr0); load_frag(k0b, Kr0 + 32);
      load_frag(k1a, Kr1); load_frag(k1b, Kr1 + 32);
      load_frag(k2a, Kr2); load_frag(k2b, Kr2 + 32);
      load_frag(k3a, Kr3); load_frag(k3b, Kr3 + 32);
      c0 = wmma_bf16(qa0, k0a, c0);
      c1 = wmma_bf16(qa0, k1a, c1);
      c2 = wmma_bf16(qa0, k2a, c2);
      c3 = wmma_bf16(qa0, k3a, c3);
      c0 = wmma_bf16(qa1, k0b, c0);
      c1 = wmma_bf16(qa1, k1b, c1);
      c2 = wmma_bf16(qa1, k2b, c2);
      c3 = wmma_bf16(qa1, k3b, c3);
    }
    // ---- V fragment loads issued EARLY: latency hidden under the softmax VALU
    Frag v0a, v0b, v1a, v1b, v2a, v2b, v3a, v3b;
    {
      const bf16* Vr0 = Vbase + ((size_t)ln << 12) + k0;
      const bf16* Vr1 = Vr0 + ((size_t)16 << 12);
      const bf16* Vr2 = Vr0 + ((size_t)32 << 12);
      const bf16* Vr3 = Vr0 + ((size_t)48 << 12);
      load_frag(v0a, Vr0); load_frag(v0b, Vr0 + 32);
      load_frag(v1a, Vr1); load_frag(v1b, Vr1 + 32);
      load_frag(v2a, Vr2); load_frag(v2b, Vr2 + 32);
      load_frag(v3a, Vr3); load_frag(v3b, Vr3 + 32);
    }
    // ---- online softmax over 64 columns (row stats per lane for rows j + 8*half)
    #pragma unroll
    for (int j = 0; j < 8; ++j) {
      float x0 = c0[j] * scale, x1 = c1[j] * scale;
      float x2 = c2[j] * scale, x3 = c3[j] * scale;
      float mx = fmaxf(fmaxf(x0, x1), fmaxf(x2, x3));
      #pragma unroll
      for (int off = 1; off < 16; off <<= 1)
        mx = fmaxf(mx, __shfl_xor(mx, off, 32));     // stays within 16-lane half
      const float mnew  = fmaxf(Mx[j], mx);
      const float alpha = __expf(Mx[j] - mnew);
      Mx[j] = mnew;
      const float p0 = __expf(x0 - mnew), p1 = __expf(x1 - mnew);
      const float p2 = __expf(x2 - mnew), p3 = __expf(x3 - mnew);
      float rs = (p0 + p1) + (p2 + p3);
      #pragma unroll
      for (int off = 1; off < 16; off <<= 1)
        rs += __shfl_xor(rs, off, 32);
      l[j] = l[j] * alpha + rs;
      O0[j] *= alpha; O1[j] *= alpha; O2[j] *= alpha; O3[j] *= alpha;
      const int row = j + half * 8;                  // C-layout row
      bf16* pl = &Plds[wave][row * 64 + ln];
      pl[0]  = to_bf16(p0);
      pl[16] = to_bf16(p1);
      pl[32] = to_bf16(p2);
      pl[48] = to_bf16(p3);
    }
    asm volatile("s_wait_dscnt 0" ::: "memory");     // LDS store -> load, same wave
    // ---- P in A-layout from LDS: keys 0..31 (pa0) and 32..63 (pa1)
    Frag pa0, pa1;
    {
      const bf16* pl = &Plds[wave][ln * 64];
      pa0.h[0] = *(const v8bf*)(pl + klo);
      pa0.h[1] = *(const v8bf*)(pl + klo + 16);
      pa1.h[0] = *(const v8bf*)(pl + 32 + klo);
      pa1.h[1] = *(const v8bf*)(pl + 32 + klo + 16);
    }
    // ---- O += P @ V  (K = 64 keys), 8 WMMAs (V already resident)
    O0 = wmma_bf16(pa0, v0a, O0);
    O1 = wmma_bf16(pa0, v1a, O1);
    O2 = wmma_bf16(pa0, v2a, O2);
    O3 = wmma_bf16(pa0, v3a, O3);
    O0 = wmma_bf16(pa1, v0b, O0);
    O1 = wmma_bf16(pa1, v1b, O1);
    O2 = wmma_bf16(pa1, v2b, O2);
    O3 = wmma_bf16(pa1, v3b, O3);
  }
  // ---- finalize: O / l
  bf16* Cr = ctx + (size_t)(b * 4096 + q0) * 512 + h * 64;
  #pragma unroll
  for (int j = 0; j < 8; ++j) {
    const int row = j + half * 8;
    const float inv = 1.0f / l[j];
    bf16* cp = Cr + (size_t)row * 512 + ln;
    cp[0]  = to_bf16(O0[j] * inv);
    cp[16] = to_bf16(O1[j] * inv);
    cp[32] = to_bf16(O2[j] * inv);
    cp[48] = to_bf16(O3[j] * inv);
  }
}

// ---------------------------------------------------------------- launcher
extern "C" void kernel_launch(void* const* d_in, const int* in_sizes, int n_in,
                              void* d_out, int out_size, void* d_ws, size_t ws_size,
                              hipStream_t stream) {
  const float* q  = (const float*)d_in[0];
  const float* k  = (const float*)d_in[1];
  const float* v  = (const float*)d_in[2];
  const float* Wq = (const float*)d_in[3];
  const float* bq = (const float*)d_in[4];
  const float* Wk = (const float*)d_in[5];
  const float* bk = (const float*)d_in[6];
  const float* Wv = (const float*)d_in[7];
  const float* bv = (const float*)d_in[8];
  const float* Wo = (const float*)d_in[9];
  const float* bo = (const float*)d_in[10];

  const int BS  = 8192;                 // B*S = 2*4096
  const size_t BSD = (size_t)BS * 512;  // 4,194,304 elems
  const int WN = 512 * 512;             // 262,144 elems

  char* ws = (char*)d_ws;
  bf16* qb  = (bf16*)(ws);                          // 8 MB
  bf16* kbf = (bf16*)(ws + 2 * BSD);                // 8 MB
  bf16* vbf = (bf16*)(ws + 4 * BSD);                // 8 MB
  bf16* Wqb = (bf16*)(ws + 6 * BSD);                // 0.5 MB each
  bf16* Wkb = Wqb + WN;
  bf16* Wvb = Wkb + WN;
  bf16* Wob = Wvb + WN;
  bf16* Qh  = Wob + WN;                             // 8 MB
  bf16* Khp = Qh + BSD;                             // 8 MB
  bf16* Vt  = Khp + BSD;                            // 8 MB
  bf16* ctx = qb;                                   // reuse: qb dead after Q proj

  const int NE4 = (int)(BSD / 4);
  const int WN4 = WN / 4;
  mha_cvt_bf16<<<(NE4 + 255) / 256, 256, 0, stream>>>(q, qb, NE4);
  mha_cvt_bf16<<<(NE4 + 255) / 256, 256, 0, stream>>>(k, kbf, NE4);
  mha_cvt_bf16<<<(NE4 + 255) / 256, 256, 0, stream>>>(v, vbf, NE4);
  mha_cvt_bf16<<<(WN4 + 255) / 256, 256, 0, stream>>>(Wq, Wqb, WN4);
  mha_cvt_bf16<<<(WN4 + 255) / 256, 256, 0, stream>>>(Wk, Wkb, WN4);
  mha_cvt_bf16<<<(WN4 + 255) / 256, 256, 0, stream>>>(Wv, Wvb, WN4);
  mha_cvt_bf16<<<(WN4 + 255) / 256, 256, 0, stream>>>(Wo, Wob, WN4);

  const int gblocks = (BS / 16) * (512 / 64) / 4;   // 1024 blocks of 4 waves
  mha_gemm_xwt<<<gblocks, 128, 0, stream>>>(qb,  Wqb, bq, Qh,  nullptr, BS, 0);
  mha_gemm_xwt<<<gblocks, 128, 0, stream>>>(kbf, Wkb, bk, Khp, nullptr, BS, 0);
  mha_gemm_xwt<<<gblocks, 128, 0, stream>>>(vbf, Wvb, bv, Vt,  nullptr, BS, 1);

  mha_flash_attn<<<512, 256, 0, stream>>>(Qh, Khp, Vt, ctx);

  mha_gemm_xwt<<<gblocks, 128, 0, stream>>>(ctx, Wob, bo, nullptr, (float*)d_out, BS, 2);
}
